// VoxelResBackBone8xVoxelNeXtEffv2Next4_84559316123982
// MI455X (gfx1250) — matrix operand order repaired and checked by reading, not compiled, not run on hardware
//
#include <hip/hip_runtime.h>
#include <hip/hip_bf16.h>
#include <math.h>

typedef __attribute__((ext_vector_type(16))) _Float16 v16h;
typedef __attribute__((ext_vector_type(8)))  float    v8f;
typedef __attribute__((ext_vector_type(4)))  float    f4;

#define CCH   128   // channels
#define BQ    8     // batches
#define HID   16    // hidden
#define NBLK  2048  // persistent grid for streaming passes

// ---------------------------------------------------------------- init
__global__ void k_init(float* __restrict__ seg, unsigned* __restrict__ cnt) {
    int t = threadIdx.x;
    for (int i = t; i < BQ * CCH; i += 256) seg[i] = 0.0f;
    if (t < BQ) cnt[t] = 0u;
}

// ---------------------------------------------------------------- pass 1: segment sums
// 256 threads: lane (t&31) owns one float4 column group, (t>>5) selects row slot.
// batch_idx is sorted -> accumulate in registers, flush to LDS atomics on batch
// change, flush LDS to global atomics once per block.
__global__ void k_reduce(const float* __restrict__ f3, const float* __restrict__ f5,
                         const int* __restrict__ bidx,
                         float* __restrict__ seg, unsigned* __restrict__ cnt, int N) {
    __shared__ float    lsum[BQ * CCH];
    __shared__ unsigned lcnt[BQ];
    const int t  = threadIdx.x;
    const int cg = t & 31;       // float4 column group 0..31
    const int rl = t >> 5;       // row slot 0..7
    for (int i = t; i < BQ * CCH; i += 256) lsum[i] = 0.0f;
    if (t < BQ) lcnt[t] = 0u;
    __syncthreads();

    f4 acc = (f4)0.0f;
    int curb = -1;
    unsigned cc = 0;
    const int stride = 8 * gridDim.x;

    auto flush = [&]() {
        if (curb >= 0) {
            float* p = &lsum[curb * CCH + (cg << 2)];
            atomicAdd(p + 0, acc[0]);
            atomicAdd(p + 1, acc[1]);
            atomicAdd(p + 2, acc[2]);
            atomicAdd(p + 3, acc[3]);
            if (cg == 0) atomicAdd(&lcnt[curb], cc);
        }
        acc = (f4)0.0f;
        cc = 0;
    };

    for (int row = blockIdx.x * 8 + rl; row < N; row += stride) {
        const int b = bidx[row];
        if (b != curb) { flush(); curb = b; }
        const long off = (long)row * CCH + (cg << 2);
        const f4 a = *(const f4*)(f3 + off);
        const f4 c = *(const f4*)(f5 + off);
        acc += a + c;
        ++cc;
    }
    flush();
    __syncthreads();

    for (int i = t; i < BQ * CCH; i += 256)
        if (lsum[i] != 0.0f) atomicAdd(&seg[i], lsum[i]);
    if (t < BQ && lcnt[t]) atomicAdd(&cnt[t], lcnt[t]);
}

// ---------------------------------------------------------------- pass 2: tiny GEMM chain (1 wave)
// z = feat_s(16x128, rows 8..15 zero) @ w_sq(128x16)  -> 4x WMMA f32_16x16x32_f16
// BN(eps=1e-3)+ReLU, then ext{3,5} = feat_z(16x16) @ w_ex(16x128) with K padded
// to 32 -> 16x WMMA; 2-way softmax; store att[8][2][128].
__global__ void k_att(const float* __restrict__ seg, const unsigned* __restrict__ cnt,
                      const float* __restrict__ w_sq,
                      const float* __restrict__ bn_g, const float* __restrict__ bn_b,
                      const float* __restrict__ bn_m, const float* __restrict__ bn_v,
                      const float* __restrict__ wx3, const float* __restrict__ wx5,
                      float* __restrict__ att) {
    __shared__ float sS[16 * CCH];  // feat_s, zero padded to 16 rows
    __shared__ float sZ[16 * HID];  // feat_z
    const int L  = threadIdx.x;     // 0..31
    const int hi = (L >= 16) ? 1 : 0;
    const int Lm = L & 15;

    for (int i = L; i < 16 * CCH; i += 32) {
        const int r = i >> 7, c = i & 127;
        float v = 0.0f;
        if (r < BQ) { const unsigned ct = cnt[r]; v = ct ? seg[r * CCH + c] / (float)ct : 0.0f; }
        sS[i] = v;
    }
    __syncthreads();

    // ---- squeeze GEMM: K = 128 in 4 chunks of 32
    v8f z = {};
    for (int kc = 0; kc < 4; ++kc) {
        const int kbase = kc * 32;
        v16h a, b;
        for (int v = 0; v < 8; ++v)
            for (int h = 0; h < 2; ++h) {
                // 16-bit A 16x32 layout: K = (v/4)*16 + hi*8 + (v%4)*2 + h, M = Lm
                const int ka = ((v >> 2) << 4) + (hi << 3) + ((v & 3) << 1) + h;
                a[v * 2 + h] = (_Float16)sS[Lm * CCH + kbase + ka];
                // B 32x16 layout: K = (v/4)*16 + hi*8 + (v%4)*2 + h, N = Lm
                const int kb = ((v >> 2) << 4) + (hi << 3) + ((v & 3) << 1) + h;
                b[v * 2 + h] = (_Float16)w_sq[(kbase + kb) * HID + Lm];
            }
        z = __builtin_amdgcn_wmma_f32_16x16x32_f16(false, a, false, b, (short)0, z, false, false);
    }

    // ---- BN (inference) + ReLU; hidden index = Lm
    const float sc = bn_g[Lm] * rsqrtf(bn_v[Lm] + 1e-3f);
    const float sh = bn_b[Lm] - bn_m[Lm] * sc;
    for (int j = 0; j < 8; ++j) {
        const float tv = fmaf(z[j], sc, sh);
        z[j] = tv > 0.0f ? tv : 0.0f;
    }
    // C/D layout: VGPR j -> (M = j + hi*8, N = Lm)
    for (int j = 0; j < 8; ++j) sZ[(j + (hi << 3)) * HID + Lm] = z[j];
    __syncthreads();

    // ---- excitation: A = feat_z padded K 16->32 (built once)
    v16h az;
    for (int v = 0; v < 8; ++v)
        for (int h = 0; h < 2; ++h) {
            const int ka = ((v >> 2) << 4) + (hi << 3) + ((v & 3) << 1) + h;
            az[v * 2 + h] = (_Float16)(ka < HID ? sZ[Lm * HID + ka] : 0.0f);
        }

    for (int nb = 0; nb < CCH; nb += 16) {
        v16h b3, b5;
        for (int v = 0; v < 8; ++v)
            for (int h = 0; h < 2; ++h) {
                const int kb = ((v >> 2) << 4) + (hi << 3) + ((v & 3) << 1) + h;
                float v3 = 0.0f, v5 = 0.0f;
                if (kb < HID) { v3 = wx3[kb * CCH + nb + Lm]; v5 = wx5[kb * CCH + nb + Lm]; }
                b3[v * 2 + h] = (_Float16)v3;
                b5[v * 2 + h] = (_Float16)v5;
            }
        v8f e3 = {}, e5 = {};
        e3 = __builtin_amdgcn_wmma_f32_16x16x32_f16(false, az, false, b3, (short)0, e3, false, false);
        e5 = __builtin_amdgcn_wmma_f32_16x16x32_f16(false, az, false, b5, (short)0, e5, false, false);
        // valid batches are M=0..7 -> VGPR j=0..7 on lanes 0..15
        if (!hi) {
            for (int j = 0; j < BQ; ++j) {
                const float x3 = e3[j], x5 = e5[j];
                const float mx = fmaxf(x3, x5);
                const float q3 = __expf(x3 - mx), q5 = __expf(x5 - mx);
                const float inv = 1.0f / (q3 + q5);
                att[(j * 2 + 0) * CCH + nb + Lm] = q3 * inv;
                att[(j * 2 + 1) * CCH + nb + Lm] = q5 * inv;
            }
        }
    }
}

// ---------------------------------------------------------------- pass 3: weighted combine
__global__ void k_combine(const float* __restrict__ f3, const float* __restrict__ f5,
                          const int* __restrict__ bidx, const float* __restrict__ att,
                          float* __restrict__ out, int N) {
    __shared__ float latt[BQ * 2 * CCH];  // 8 KB, hot
    const int t = threadIdx.x;
    for (int i = t; i < BQ * 2 * CCH; i += 256) latt[i] = att[i];
    __syncthreads();

    const int cg = t & 31;
    const int rl = t >> 5;
    const int stride = 8 * gridDim.x;
    for (int row = blockIdx.x * 8 + rl; row < N; row += stride) {
        if (row + stride < N) {
            __builtin_prefetch(f3 + (long)(row + stride) * CCH + (cg << 2), 0, 1);
            __builtin_prefetch(f5 + (long)(row + stride) * CCH + (cg << 2), 0, 1);
        }
        const int b = bidx[row];
        const f4 a3 = *(const f4*)&latt[(b * 2 + 0) * CCH + (cg << 2)];
        const f4 a5 = *(const f4*)&latt[(b * 2 + 1) * CCH + (cg << 2)];
        const long off = (long)row * CCH + (cg << 2);
        const f4 x3 = *(const f4*)(f3 + off);
        const f4 x5 = *(const f4*)(f5 + off);
        f4 o;
        o[0] = fmaf(x3[0], a3[0], x5[0] * a5[0]);
        o[1] = fmaf(x3[1], a3[1], x5[1] * a5[1]);
        o[2] = fmaf(x3[2], a3[2], x5[2] * a5[2]);
        o[3] = fmaf(x3[3], a3[3], x5[3] * a5[3]);
        __builtin_nontemporal_store(o, (f4*)(out + off));  // output is never re-read
    }
}

// ---------------------------------------------------------------- launch
extern "C" void kernel_launch(void* const* d_in, const int* in_sizes, int n_in,
                              void* d_out, int out_size, void* d_ws, size_t ws_size,
                              hipStream_t stream) {
    const float* f3   = (const float*)d_in[0];
    const float* f5   = (const float*)d_in[1];
    const int*   bidx = (const int*)d_in[2];
    const float* w_sq = (const float*)d_in[3];
    const float* bn_g = (const float*)d_in[4];
    const float* bn_b = (const float*)d_in[5];
    const float* bn_m = (const float*)d_in[6];
    const float* bn_v = (const float*)d_in[7];
    const float* wx3  = (const float*)d_in[8];
    const float* wx5  = (const float*)d_in[9];
    float* out = (float*)d_out;

    const int N = in_sizes[0] / CCH;

    char* ws = (char*)d_ws;
    float*    seg = (float*)(ws);             // 1024 floats
    unsigned* cnt = (unsigned*)(ws + 4096);   // 8 u32
    float*    att = (float*)(ws + 8192);      // 2048 floats

    k_init<<<1, 256, 0, stream>>>(seg, cnt);
    k_reduce<<<NBLK, 256, 0, stream>>>(f3, f5, bidx, seg, cnt, N);
    k_att<<<1, 32, 0, stream>>>(seg, cnt, w_sq, bn_g, bn_b, bn_m, bn_v, wx3, wx5, att);
    k_combine<<<NBLK, 256, 0, stream>>>(f3, f5, bidx, att, out, N);
}
